// Large_batch_queue_classwise_46548855554601
// MI455X (gfx1250) — matrix operand order repaired and checked by reading, not compiled, not run on hardware
//
#include <hip/hip_runtime.h>

// CDNA5 (gfx1250) segment-mean via one-hot WMMA GEMM.
//   sums[48,256] = onehot(labels)^T[48,N] @ features[N,256]
// f16 WMMA with hi+lo residual split => ~f32 accuracy, f32 accumulate.

typedef __attribute__((ext_vector_type(16))) _Float16 v16h;
typedef __attribute__((ext_vector_type(8)))  float    v8f;

#define NCLS  37
#define CPAD  48      // 3 class tiles x 16
#define FEAT  256
#define NB    256     // number of partial-sum blocks
#define BLK   512     // 16 waves of 32
#define CHROWS 32     // rows (K) per chunk

__global__ __launch_bounds__(BLK)
void seg_wmma_kernel(const float* __restrict__ feat,
                     const int*   __restrict__ lab64,   // int64 labels, view as int pairs
                     float*       __restrict__ sums_part, // [NB][CPAD][FEAT]
                     float*       __restrict__ cnts_part, // [NB][CPAD]
                     int nRows)
{
    const int tid     = threadIdx.x;
    const int lane    = tid & 31;
    const int wv      = tid >> 5;        // colTile 0..15
    const int half    = lane >> 4;       // lane half (WMMA fragment layout)
    const int nn      = lane & 15;       // N index within tile
    const int colBase = wv * 16;
    const int b       = blockIdx.x;

    v8f   acc[3] = {};
    float cnt[3] = {0.f, 0.f, 0.f};

    const int nChunks = nRows >> 5;
    for (int chunk = b; chunk < nChunks; chunk += NB) {
        const int rowBase = chunk << 5;

        // one label per lane (low dword of int64), then shuffle into the
        // A-fragment K order: lane half h holds K = h*16+e
        int lab = lab64[(size_t)(rowBase + lane) * 2];
        int labv[16];
#pragma unroll
        for (int e = 0; e < 16; ++e) {
            int k = half * 16 + e;   // A-matrix K(e)
            labv[e] = __shfl(lab, k, 32);
        }

        // B fragment (32x16 f16): lanes 0-15 K=0..15, lanes 16-31 K=16..31
        // lane reads column (colBase+nn) over 16 rows; 64B coalesced per half wave
        const float* fp = feat + (size_t)(rowBase + half * 16) * FEAT + colBase + nn;
        float vals[16];
#pragma unroll
        for (int e = 0; e < 16; ++e)
            vals[e] = fp[(size_t)e * FEAT];

        // prefetch next chunk of this lane's stream (global_prefetch_b8)
        if (chunk + NB < nChunks)
            __builtin_prefetch(fp + (size_t)NB * CHROWS * FEAT, 0, 0);

        // split f32 -> f16 hi + f16 residual lo (keeps ~f32 accuracy)
        v16h bhi, blo;
#pragma unroll
        for (int e = 0; e < 16; ++e) {
            float v = vals[e];
            _Float16 h = (_Float16)v;
            bhi[e] = h;
            blo[e] = (_Float16)(v - (float)h);
        }

        // three class tiles: one-hot A (exact in f16) + 2 chained WMMAs each
#pragma unroll
        for (int ct = 0; ct < 3; ++ct) {
            const int cls = ct * 16 + nn;
            v16h a;
#pragma unroll
            for (int e = 0; e < 16; ++e)
                a[e] = (labv[e] == cls) ? (_Float16)1.0f : (_Float16)0.0f;

            acc[ct] = __builtin_amdgcn_wmma_f32_16x16x32_f16(
                false, a, false, bhi, (short)0, acc[ct], false, false);
            acc[ct] = __builtin_amdgcn_wmma_f32_16x16x32_f16(
                false, a, false, blo, (short)0, acc[ct], false, false);

            if (wv == 0) {                 // wave 0 also tallies class counts
                float cc = 0.f;
#pragma unroll
                for (int e = 0; e < 16; ++e)
                    cc += (labv[e] == cls) ? 1.0f : 0.0f;
                cnt[ct] += cc;
            }
        }
    }

    // write per-block partial sums: D layout -> class = ct*16 + r + 8*half
    const size_t base = (size_t)b * CPAD * FEAT;
#pragma unroll
    for (int ct = 0; ct < 3; ++ct) {
#pragma unroll
        for (int r = 0; r < 8; ++r) {
            int c = ct * 16 + r + 8 * half;
            sums_part[base + (size_t)c * FEAT + colBase + nn] = acc[ct][r];
        }
    }

    // counts: lane m (half0) + lane m+16 (half1) cover all 32 rows of a chunk
    if (wv == 0) {
#pragma unroll
        for (int ct = 0; ct < 3; ++ct) {
            float tot = cnt[ct] + __shfl_xor(cnt[ct], 16, 32);
            if (half == 0)
                cnts_part[(size_t)b * CPAD + ct * 16 + nn] = tot;
        }
    }
}

__global__ __launch_bounds__(256)
void finalize_kernel(const float* __restrict__ sums_part,
                     const float* __restrict__ cnts_part,
                     const float* __restrict__ queue,   // [37][2][256]
                     const int*   __restrict__ tail64,  // int64 per class
                     float*       __restrict__ out)     // [37][2][256]
{
    const int c = blockIdx.x;     // 0..36
    const int f = threadIdx.x;    // 0..255

    float s = 0.f;
    for (int bb = 0; bb < NB; ++bb)
        s += sums_part[(size_t)bb * CPAD * FEAT + (size_t)c * FEAT + f];

    __shared__ float sh[256];
    float cv = 0.f;
    for (int bb = f; bb < NB; bb += 256)
        cv += cnts_part[(size_t)bb * CPAD + c];
    sh[f] = cv;
    __syncthreads();
    for (int st = 128; st > 0; st >>= 1) {
        if (f < st) sh[f] += sh[f + st];
        __syncthreads();
    }
    const float count = sh[0];

    int slot = tail64[2 * c];                 // low dword of int64
    slot = slot < 0 ? 0 : (slot > 1 ? 1 : slot);
    const int other = 1 - slot;

    const float mean = s / fmaxf(count, 1.0f);
    const float* q = queue + (size_t)c * 2 * FEAT;
    float*       o = out   + (size_t)c * 2 * FEAT;

    o[(size_t)slot  * FEAT + f] = (count > 0.f) ? mean : q[(size_t)slot * FEAT + f];
    o[(size_t)other * FEAT + f] = q[(size_t)other * FEAT + f];
}

extern "C" void kernel_launch(void* const* d_in, const int* in_sizes, int n_in,
                              void* d_out, int out_size, void* d_ws, size_t ws_size,
                              hipStream_t stream)
{
    const float* feat  = (const float*)d_in[0];   // [N,256] f32
    const int*   lab   = (const int*)  d_in[1];   // [N] int64
    const float* queue = (const float*)d_in[2];   // [37,2,256] f32
    const int*   tail  = (const int*)  d_in[3];   // [37] int64
    float*       out   = (float*)d_out;           // [37,2,256] f32

    const int N = in_sizes[0] / FEAT;

    float* sums_part = (float*)d_ws;                               // NB*48*256 f32
    float* cnts_part = sums_part + (size_t)NB * CPAD * FEAT;       // NB*48 f32

    seg_wmma_kernel<<<NB, BLK, 0, stream>>>(feat, lab, sums_part, cnts_part, N);
    finalize_kernel<<<NCLS, 256, 0, stream>>>(sums_part, cnts_part, queue, tail, out);
}
